// AttentionBlock_60601988547057
// MI455X (gfx1250) — matrix operand order repaired and checked by reading, not compiled, not run on hardware
//
#include <hip/hip_runtime.h>
#include <hip/hip_bf16.h>

// ---------------------------------------------------------------------------
// Problem dimensions (from reference)
// ---------------------------------------------------------------------------
constexpr int NB  = 2;      // batch
constexpr int CD  = 256;    // channels
constexpr int SD  = 64 * 64;// sequence length (w*h) = 4096
constexpr int NH  = 4;      // heads
constexpr int DHD = CD / NH;// per-head dim = 64
constexpr float EPS   = 1e-5f;
constexpr float SCALE = 0.015625f; // 1/sqrt(4096)

// ---------------------------------------------------------------------------
// WMMA types (CDNA5 gfx1250, wave32)
// ---------------------------------------------------------------------------
typedef __attribute__((ext_vector_type(16))) __bf16          v16bf;
typedef __attribute__((ext_vector_type(8)))  float           v8f;
typedef __attribute__((ext_vector_type(8)))  unsigned short  us8;

union BF16Frag { v16bf v; unsigned short u[16]; us8 h[2]; };

__device__ inline unsigned short f2bf(float f) {
    union { float f; unsigned u; } x; x.f = f;
    unsigned u = x.u;
    unsigned r = (u + 0x7FFFu + ((u >> 16) & 1u)) >> 16;   // round-nearest-even
    return (unsigned short)r;
}

__device__ inline v8f wmma_bf16(v16bf a, v16bf b, v8f c) {
    // D = A(16x32 bf16) * B(32x16 bf16) + C(16x16 f32)
    return __builtin_amdgcn_wmma_f32_16x16x32_bf16(
        /*neg_a=*/false, a, /*neg_b=*/false, b,
        /*c_mod=*/(short)0, c, /*reuse_a=*/false, /*reuse_b=*/false);
}

// A fragment: 16x32 tile of row-major bf16 matrix, origin (row0, k0).
// Lane L: m = L%16 ; low lanes take K {0-7,16-23}, high lanes K {8-15,24-31}.
// Per lane: two contiguous 16-byte chunks -> global_load_b128 x2.
__device__ inline v16bf load_a(const unsigned short* A, int ld, int row0, int k0) {
    int lane = threadIdx.x & 31;
    int m    = lane & 15;
    int kb   = (lane >> 4) * 8;
    const unsigned short* p = A + (size_t)(row0 + m) * ld + k0 + kb;
    BF16Frag fr;
    fr.h[0] = *(const us8*)(p);        // K = kb .. kb+7
    fr.h[1] = *(const us8*)(p + 16);   // K = 16+kb .. 16+kb+7
    return fr.v;
}

// B fragment for logical B = M^T of a row-major matrix M (B[k,n] = M[col0+n, k]).
// Lane L: n = L%16 ; low lanes K {0-7,16-23}, high lanes K {8-15,24-31} —
// two contiguous 16-byte chunks per lane -> global_load_b128 x2.
__device__ inline v16bf load_bt(const unsigned short* M, int ld, int k0, int col0) {
    int lane = threadIdx.x & 31;
    int n    = lane & 15;
    int kb   = (lane >> 4) * 8;
    const unsigned short* p = M + (size_t)(col0 + n) * ld + k0 + kb;
    BF16Frag fr;
    fr.h[0] = *(const us8*)(p);
    fr.h[1] = *(const us8*)(p + 16);
    return fr.v;
}

// ---------------------------------------------------------------------------
// f32 -> bf16 with transpose: out[b, c, r] = bf16(in[b, r, c])
// ---------------------------------------------------------------------------
__global__ void cvt_bf16_t_kernel(const float* __restrict__ in,
                                  unsigned short* __restrict__ out,
                                  int batch, int rows, int cols) {
    int i = blockIdx.x * blockDim.x + threadIdx.x;
    int total = batch * rows * cols;
    if (i >= total) return;
    int c = i % cols;
    int t = i / cols;
    int r = t % rows;
    int b = t / rows;
    out[((size_t)b * cols + c) * rows + r] = f2bf(in[i]);
}

// ---------------------------------------------------------------------------
// K0: transpose x[N,C,S] -> r[N,S,C] and LayerNorm1 -> norm (bf16).
// Tiled through LDS: block handles 16 s-positions; global reads/writes are
// contiguous 64B segments per 16-lane group (no per-lane 16KB strides).
// ---------------------------------------------------------------------------
__global__ __launch_bounds__(256)
void ln1_kernel(const float* __restrict__ x,
                const float* __restrict__ w, const float* __restrict__ b,
                float* __restrict__ r, unsigned short* __restrict__ normbf) {
    __shared__ float tile[16][CD + 1];
    __shared__ float part[16][16];
    __shared__ float mu_s[16], rs_s[16];

    int blk = blockIdx.x;                  // NB * SD/16
    int n   = blk / (SD / 16);
    int s0  = (blk % (SD / 16)) * 16;
    int t   = threadIdx.x;
    int ts  = t & 15;                      // s offset within tile
    int tg  = t >> 4;                      // 16-lane group id

    // load: for each channel group, 16 contiguous floats along s per group
#pragma unroll
    for (int c0 = 0; c0 < CD; c0 += 16) {
        int c = c0 + tg;
        tile[ts][c] = x[((size_t)n * CD + c) * SD + s0 + ts];
    }
    __syncthreads();

    // per-s statistics: thread (sl = tg, cg = ts) strides channels by 16
    int sl = tg, cg = ts;
    float s1 = 0.f, s2 = 0.f;
#pragma unroll
    for (int c = cg; c < CD; c += 16) {
        float v = tile[sl][c];
        s1 += v; s2 += v * v;
    }
    part[sl][cg] = s1;
    __syncthreads();
    if (cg == 0) {
        float a = 0.f;
#pragma unroll
        for (int j = 0; j < 16; ++j) a += part[sl][j];
        mu_s[sl] = a * (1.0f / CD);
    }
    __syncthreads();
    part[sl][cg] = s2;
    __syncthreads();
    if (cg == 0) {
        float a = 0.f;
#pragma unroll
        for (int j = 0; j < 16; ++j) a += part[sl][j];
        float mu = mu_s[sl];
        rs_s[sl] = rsqrtf(a * (1.0f / CD) - mu * mu + EPS);
    }
    __syncthreads();

    // write r and norm (coalesced along c)
#pragma unroll
    for (int c0 = 0; c0 < CD; c0 += 16) {
        int c = c0 + cg;
        float v = tile[sl][c];
        size_t idx = ((size_t)n * SD + s0 + sl) * CD + c;
        r[idx] = v;
        normbf[idx] = f2bf((v - mu_s[sl]) * rs_s[sl] * w[c] + b[c]);
    }
}

// ---------------------------------------------------------------------------
// K1: QKV projections. One wave per 16x16 output tile.
// Q,K stored row-major [S,DH]; V stored TRANSPOSED [DH,S] so the attention
// kernel's B-fragments of V are contiguous along K.
// ---------------------------------------------------------------------------
__global__ __launch_bounds__(256)
void qkv_kernel(const unsigned short* __restrict__ normbf,
                const unsigned short* __restrict__ Wqt,  // [H, DH, C]
                const unsigned short* __restrict__ Wkt,
                const unsigned short* __restrict__ Wvt,
                unsigned short* __restrict__ Q,
                unsigned short* __restrict__ K,
                unsigned short* __restrict__ Vt) {
    constexpr int TD = DHD / 16, TS = SD / 16;
    int t = (blockIdx.x * blockDim.x + threadIdx.x) >> 5;
    int dt = t % TD; t /= TD;
    int st = t % TS; t /= TS;
    int h  = t % NH; t /= NH;
    int n  = t % NB; t /= NB;
    int qkv = t;
    if (qkv >= 3) return;

    const unsigned short* Wsel = (qkv == 0) ? Wqt : (qkv == 1) ? Wkt : Wvt;
    const unsigned short* Ab = normbf + (size_t)n * SD * CD;
    const unsigned short* Bb = Wsel   + (size_t)h * DHD * CD;

    v8f acc = {};
#pragma unroll
    for (int kc = 0; kc < CD; kc += 32) {
        v16bf a = load_a(Ab, CD, st * 16, kc);
        v16bf b = load_bt(Bb, CD, kc, dt * 16);   // B[k,n] = Wt[h, d, k]
        acc = wmma_bf16(a, b, acc);
    }

    int lane = threadIdx.x & 31;
    int cn   = lane & 15;
    int mh   = (lane >> 4) * 8;
    if (qkv == 2) {
        unsigned short* ob = Vt + (size_t)(n * NH + h) * SD * DHD;
#pragma unroll
        for (int i = 0; i < 8; ++i) {
            int s = st * 16 + i + mh;
            ob[(size_t)(dt * 16 + cn) * SD + s] = f2bf(acc[i]);   // Vt[d, s]
        }
    } else {
        unsigned short* ob = ((qkv == 0) ? Q : K) + (size_t)(n * NH + h) * SD * DHD;
#pragma unroll
        for (int i = 0; i < 8; ++i) {
            int m = i + mh;
            ob[(size_t)(st * 16 + m) * DHD + dt * 16 + cn] = f2bf(acc[i]);
        }
    }
}

// ---------------------------------------------------------------------------
// K2: column softmax statistics (softmax is over the QUERY axis).
// One wave per 16-key column block; key B-fragments hoisted (loop-invariant),
// streams all query tiles with online max / sum-exp.
// ---------------------------------------------------------------------------
__global__ __launch_bounds__(256)
void colstats_kernel(const unsigned short* __restrict__ Q,
                     const unsigned short* __restrict__ K,
                     float* __restrict__ mcol, float* __restrict__ lcol) {
    constexpr int TS = SD / 16;
    int t = (blockIdx.x * blockDim.x + threadIdx.x) >> 5;
    int kt = t % TS; t /= TS;
    int h  = t % NH; t /= NH;
    int n  = t;
    if (n >= NB) return;

    const unsigned short* Qb = Q + (size_t)(n * NH + h) * SD * DHD;
    const unsigned short* Kb = K + (size_t)(n * NH + h) * SD * DHD;

    // key-block fragments are invariant over the query loop
    v16bf bk0 = load_bt(Kb, DHD, 0,  kt * 16);
    v16bf bk1 = load_bt(Kb, DHD, 32, kt * 16);

    int lane = threadIdx.x & 31;
    float run_m = -3.0e38f, run_l = 0.0f;

    for (int qt = 0; qt < TS; ++qt) {
        v8f acc = {};
        acc = wmma_bf16(load_a(Qb, DHD, qt * 16, 0),  bk0, acc);
        acc = wmma_bf16(load_a(Qb, DHD, qt * 16, 32), bk1, acc);

        float tmax = -3.0e38f;
#pragma unroll
        for (int i = 0; i < 8; ++i) tmax = fmaxf(tmax, acc[i] * SCALE);
        tmax = fmaxf(tmax, __shfl_xor(tmax, 16, 32)); // merge both m-halves
        float nm = fmaxf(run_m, tmax);
        float ps = 0.0f;
#pragma unroll
        for (int i = 0; i < 8; ++i) ps += __expf(acc[i] * SCALE - nm);
        ps += __shfl_xor(ps, 16, 32);
        run_l = run_l * __expf(run_m - nm) + ps;
        run_m = nm;
    }
    if (lane < 16) {
        size_t base = (size_t)(n * NH + h) * SD + kt * 16 + lane;
        mcol[base] = run_m;
        lcol[base] = run_l;
    }
}

// ---------------------------------------------------------------------------
// K3: attention output. attn[q,d] = sum_k exp(Y[q,k]-m_k)/l_k * V[k,d].
// One wave per 16(q) x 64(ALL d) output strip: per 32-key step, Y is computed
// ONCE, softmax'd into LDS once, and the P fragment feeds FOUR WMMAs against
// the four V d-tiles (Y-recompute amortized 4x vs per-(q,d)-tile waves).
// ---------------------------------------------------------------------------
__global__ __launch_bounds__(256)
void attn_kernel(const unsigned short* __restrict__ Q,
                 const unsigned short* __restrict__ K,
                 const unsigned short* __restrict__ Vt,   // [DH, S]
                 const float* __restrict__ mcol, const float* __restrict__ lcol,
                 float* __restrict__ attn_cat) {
    constexpr int TD = DHD / 16, TS = SD / 16;
    int t = (blockIdx.x * blockDim.x + threadIdx.x) >> 5;
    int qt = t % TS; t /= TS;
    int h  = t % NH; t /= NH;
    int n  = t;
    if (n >= NB) return;

    const unsigned short* Qb  = Q  + (size_t)(n * NH + h) * SD * DHD;
    const unsigned short* Kb  = K  + (size_t)(n * NH + h) * SD * DHD;
    const unsigned short* Vb  = Vt + (size_t)(n * NH + h) * SD * DHD;
    size_t sbase = (size_t)(n * NH + h) * SD;

    __shared__ float Pldg[8][16][36];       // per-wave staging, 16B-aligned rows
    float (*P)[36] = Pldg[threadIdx.x >> 5];

    int lane = threadIdx.x & 31;
    int cn   = lane & 15;
    int mh   = (lane >> 4) * 8;

    // Q-tile fragments are invariant over the key loop
    v16bf aq0 = load_a(Qb, DHD, qt * 16, 0);
    v16bf aq1 = load_a(Qb, DHD, qt * 16, 32);

    v8f oacc[TD];
#pragma unroll
    for (int d = 0; d < TD; ++d) oacc[d] = (v8f){};

    for (int kb = 0; kb < SD; kb += 32) {
#pragma unroll
        for (int sub = 0; sub < 2; ++sub) {            // two 16-key subtiles
            v8f acc = {};
            acc = wmma_bf16(aq0, load_bt(Kb, DHD, 0,  kb + sub * 16), acc);
            acc = wmma_bf16(aq1, load_bt(Kb, DHD, 32, kb + sub * 16), acc);

            int col = kb + sub * 16 + cn;
            float mm  = mcol[sbase + col];
            float inv = 1.0f / lcol[sbase + col];
#pragma unroll
            for (int i = 0; i < 8; ++i)
                P[i + mh][sub * 16 + cn] = __expf(acc[i] * SCALE - mm) * inv;
        }
        __builtin_amdgcn_wave_barrier();               // order LDS store -> load

        // A fragment of P (16 q rows x 32 keys), f32 -> bf16, vector LDS reads
        BF16Frag fa;
        {
            int m  = lane & 15;
            int ka = (lane >> 4) * 8;
            const float* pr = P[m];
            float4 c0 = *(const float4*)(pr + ka);
            float4 c1 = *(const float4*)(pr + ka + 4);
            float4 c2 = *(const float4*)(pr + 16 + ka);
            float4 c3 = *(const float4*)(pr + 20 + ka);
            fa.u[0]  = f2bf(c0.x); fa.u[1]  = f2bf(c0.y);
            fa.u[2]  = f2bf(c0.z); fa.u[3]  = f2bf(c0.w);
            fa.u[4]  = f2bf(c1.x); fa.u[5]  = f2bf(c1.y);
            fa.u[6]  = f2bf(c1.z); fa.u[7]  = f2bf(c1.w);
            fa.u[8]  = f2bf(c2.x); fa.u[9]  = f2bf(c2.y);
            fa.u[10] = f2bf(c2.z); fa.u[11] = f2bf(c2.w);
            fa.u[12] = f2bf(c3.x); fa.u[13] = f2bf(c3.y);
            fa.u[14] = f2bf(c3.z); fa.u[15] = f2bf(c3.w);
        }
        // one P fragment drives all four V d-tiles
#pragma unroll
        for (int d = 0; d < TD; ++d) {
            v16bf bv = load_bt(Vb, SD, kb, d * 16);    // B[k,n] = Vt[d, k]
            oacc[d] = wmma_bf16(fa.v, bv, oacc[d]);
        }
        __builtin_amdgcn_wave_barrier();               // before P is reused
    }

#pragma unroll
    for (int d = 0; d < TD; ++d) {
#pragma unroll
        for (int i = 0; i < 8; ++i) {
            int m = i + mh;
            attn_cat[((size_t)n * SD + qt * 16 + m) * CD + h * DHD + d * 16 + cn] =
                oacc[d][i];
        }
    }
}

// ---------------------------------------------------------------------------
// K4: residual add + LayerNorm2 -> hbf (bf16). attn_cat kept for final add.
// (reads/writes are [n,s,c] row-major -> already coalesced)
// ---------------------------------------------------------------------------
__global__ __launch_bounds__(256)
void ln2_kernel(const float* __restrict__ attn_cat, const float* __restrict__ r,
                const float* __restrict__ w, const float* __restrict__ b,
                unsigned short* __restrict__ hbf) {
    int ns = blockIdx.x;
    int c  = threadIdx.x;
    size_t idx = (size_t)ns * CD + c;
    float v = attn_cat[idx] + r[idx];

    __shared__ float s1[256], s2[256];
    s1[c] = v; s2[c] = v * v;
    __syncthreads();
#pragma unroll
    for (int st = 128; st > 0; st >>= 1) {
        if (c < st) { s1[c] += s1[c + st]; s2[c] += s2[c + st]; }
        __syncthreads();
    }
    float mu  = s1[0] * (1.0f / CD);
    float var = s2[0] * (1.0f / CD) - mu * mu;
    float nv  = (v - mu) * rsqrtf(var + EPS) * w[c] + b[c];
    hbf[idx] = f2bf(nv);
}

// ---------------------------------------------------------------------------
// K5: MLP layer 1: g = gelu_exact(h @ W1 + b1) -> bf16  (W1 pre-transposed)
// ---------------------------------------------------------------------------
__global__ __launch_bounds__(256)
void mlp1_kernel(const unsigned short* __restrict__ hbf,
                 const unsigned short* __restrict__ W1t,
                 const float* __restrict__ b1,
                 unsigned short* __restrict__ gbf) {
    constexpr int TC = CD / 16, TS = SD / 16;
    int t = (blockIdx.x * blockDim.x + threadIdx.x) >> 5;
    int ct = t % TC; t /= TC;
    int st = t % TS; t /= TS;
    int n  = t;
    if (n >= NB) return;

    const unsigned short* Ab = hbf + (size_t)n * SD * CD;
    v8f acc = {};
#pragma unroll
    for (int kc = 0; kc < CD; kc += 32) {
        v16bf a = load_a(Ab, CD, st * 16, kc);
        v16bf b = load_bt(W1t, CD, kc, ct * 16);
        acc = wmma_bf16(a, b, acc);
    }
    int lane = threadIdx.x & 31;
    int cn   = lane & 15;
    int mh   = (lane >> 4) * 8;
    int c    = ct * 16 + cn;
    float bias = b1[c];
    unsigned short* ob = gbf + (size_t)n * SD * CD;
#pragma unroll
    for (int i = 0; i < 8; ++i) {
        float x = acc[i] + bias;
        float g = 0.5f * x * (1.0f + erff(x * 0.70710678118654752f)); // exact gelu
        ob[(size_t)(st * 16 + i + mh) * CD + c] = f2bf(g);
    }
}

// ---------------------------------------------------------------------------
// K6: MLP layer 2 + final residual + transpose back to [N,C,W,H]
// out[n,c,s] = (g @ W2 + b2)[n,s,c] + attn_cat[n,s,c]   (W2 pre-transposed)
// ---------------------------------------------------------------------------
__global__ __launch_bounds__(256)
void mlp2_kernel(const unsigned short* __restrict__ gbf,
                 const unsigned short* __restrict__ W2t,
                 const float* __restrict__ b2,
                 const float* __restrict__ attn_cat,
                 float* __restrict__ out) {
    constexpr int TC = CD / 16, TS = SD / 16;
    int t = (blockIdx.x * blockDim.x + threadIdx.x) >> 5;
    int ct = t % TC; t /= TC;
    int st = t % TS; t /= TS;
    int n  = t;
    if (n >= NB) return;

    const unsigned short* Ab = gbf + (size_t)n * SD * CD;
    v8f acc = {};
#pragma unroll
    for (int kc = 0; kc < CD; kc += 32) {
        v16bf a = load_a(Ab, CD, st * 16, kc);
        v16bf b = load_bt(W2t, CD, kc, ct * 16);
        acc = wmma_bf16(a, b, acc);
    }
    int lane = threadIdx.x & 31;
    int cn   = lane & 15;
    int mh   = (lane >> 4) * 8;
    int c    = ct * 16 + cn;
    float bias = b2[c];
#pragma unroll
    for (int i = 0; i < 8; ++i) {
        int s = st * 16 + i + mh;
        float v = acc[i] + bias + attn_cat[((size_t)n * SD + s) * CD + c];
        out[((size_t)n * CD + c) * SD + s] = v;
    }
}

// ---------------------------------------------------------------------------
// Host-side launch
// ---------------------------------------------------------------------------
extern "C" void kernel_launch(void* const* d_in, const int* in_sizes, int n_in,
                              void* d_out, int out_size, void* d_ws, size_t ws_size,
                              hipStream_t stream) {
    (void)in_sizes; (void)n_in; (void)out_size; (void)ws_size;

    const float* x    = (const float*)d_in[0];
    const float* ln1w = (const float*)d_in[1];
    const float* ln1b = (const float*)d_in[2];
    const float* WQ   = (const float*)d_in[3];
    const float* WK   = (const float*)d_in[4];
    const float* WV   = (const float*)d_in[5];
    const float* ln2w = (const float*)d_in[6];
    const float* ln2b = (const float*)d_in[7];
    const float* W1   = (const float*)d_in[8];
    const float* b1   = (const float*)d_in[9];
    const float* W2   = (const float*)d_in[10];
    const float* b2   = (const float*)d_in[11];
    float* out = (float*)d_out;

    char* wsp = (char*)d_ws;
    size_t off = 0;
    auto alloc = [&](size_t bytes) -> void* {
        void* p = wsp + off;
        off += (bytes + 255) & ~(size_t)255;
        return p;
    };

    float*          r_f32    = (float*)alloc((size_t)NB * SD * CD * 4);
    unsigned short* norm_bf  = (unsigned short*)alloc((size_t)NB * SD * CD * 2);
    unsigned short* WQt_bf   = (unsigned short*)alloc((size_t)NH * CD * DHD * 2);
    unsigned short* WKt_bf   = (unsigned short*)alloc((size_t)NH * CD * DHD * 2);
    unsigned short* WVt_bf   = (unsigned short*)alloc((size_t)NH * CD * DHD * 2);
    unsigned short* W1t_bf   = (unsigned short*)alloc((size_t)CD * CD * 2);
    unsigned short* W2t_bf   = (unsigned short*)alloc((size_t)CD * CD * 2);
    unsigned short* Q_bf     = (unsigned short*)alloc((size_t)NB * NH * SD * DHD * 2);
    unsigned short* K_bf     = (unsigned short*)alloc((size_t)NB * NH * SD * DHD * 2);
    unsigned short* Vt_bf    = (unsigned short*)alloc((size_t)NB * NH * SD * DHD * 2);
    float*          mcol     = (float*)alloc((size_t)NB * NH * SD * 4);
    float*          lcol     = (float*)alloc((size_t)NB * NH * SD * 4);
    float*          attn_cat = (float*)alloc((size_t)NB * SD * CD * 4);
    unsigned short* h_bf     = (unsigned short*)alloc((size_t)NB * SD * CD * 2);
    unsigned short* g_bf     = (unsigned short*)alloc((size_t)NB * SD * CD * 2);

    // transposed weight conversions f32 -> bf16
    int nw = NH * CD * DHD;
    cvt_bf16_t_kernel<<<(nw + 255) / 256, 256, 0, stream>>>(WQ, WQt_bf, NH, CD, DHD);
    cvt_bf16_t_kernel<<<(nw + 255) / 256, 256, 0, stream>>>(WK, WKt_bf, NH, CD, DHD);
    cvt_bf16_t_kernel<<<(nw + 255) / 256, 256, 0, stream>>>(WV, WVt_bf, NH, CD, DHD);
    int nc = CD * CD;
    cvt_bf16_t_kernel<<<(nc + 255) / 256, 256, 0, stream>>>(W1, W1t_bf, 1, CD, CD);
    cvt_bf16_t_kernel<<<(nc + 255) / 256, 256, 0, stream>>>(W2, W2t_bf, 1, CD, CD);

    // transpose + LN1 (LDS-tiled)
    ln1_kernel<<<NB * (SD / 16), 256, 0, stream>>>(x, ln1w, ln1b, r_f32, norm_bf);

    // QKV projections
    {
        int waves = 3 * NB * NH * (SD / 16) * (DHD / 16);
        qkv_kernel<<<waves / 8, 256, 0, stream>>>(norm_bf, WQt_bf, WKt_bf, WVt_bf,
                                                  Q_bf, K_bf, Vt_bf);
    }
    // column softmax statistics
    {
        int waves = NB * NH * (SD / 16);
        colstats_kernel<<<waves / 8, 256, 0, stream>>>(Q_bf, K_bf, mcol, lcol);
    }
    // attention output (one wave per 16q x 64d strip)
    {
        int waves = NB * NH * (SD / 16);
        attn_kernel<<<waves / 8, 256, 0, stream>>>(Q_bf, K_bf, Vt_bf, mcol, lcol,
                                                   attn_cat);
    }
    // residual + LN2
    ln2_kernel<<<NB * SD, 256, 0, stream>>>(attn_cat, r_f32, ln2w, ln2b, h_bf);

    // MLP
    {
        int waves = NB * (SD / 16) * (CD / 16);
        mlp1_kernel<<<waves / 8, 256, 0, stream>>>(h_bf, W1t_bf, b1, g_bf);
        mlp2_kernel<<<waves / 8, 256, 0, stream>>>(g_bf, W2t_bf, b2, attn_cat, out);
    }
}
